// ToRGBLayer_32006096290039
// MI455X (gfx1250) — matrix operand order repaired and checked by reading, not compiled, not run on hardware
//
#include <hip/hip_runtime.h>
#include <hip/hip_bf16.h>
#include <stdint.h>

typedef __attribute__((ext_vector_type(16))) __bf16 v16bf;
typedef __attribute__((ext_vector_type(8)))  float  v8f;
typedef uint32_t u32x4 __attribute__((ext_vector_type(4)));
typedef uint32_t u32x8 __attribute__((ext_vector_type(8)));

#define BATCH   8
#define W_DIM   512
#define IN_CH   256
#define OUT_CH  256
#define IN_SIZE 128
#define NPIX    (IN_SIZE*IN_SIZE)     /* 16384 */
#define CONV_OUT 130
#define PCONV   (CONV_OUT*CONV_OUT)   /* 16900 */

/* workspace byte offsets (16B aligned) */
#define WS_SN     0u
#define WS_DCOEF  8192u
#define WS_WSQ    16384u      /* 256*256 f32            */
#define WS_WQ     278528u     /* 9*256*256 bf16 swizzled */
#define WS_XT     1458176u    /* [b][pix][ci] bf16       */
#define WS_Y      68567040u   /* [b][o][130*130] f32     */

static __device__ __forceinline__ __bf16 f2bf(float f) {
    union { float f; uint32_t u; } v; v.f = f;
    uint32_t r = (v.u + 0x7FFFu + ((v.u >> 16) & 1u)) >> 16;
    union { uint16_t u; __bf16 b; } o; o.u = (uint16_t)r;
    return o.b;
}

/* ---- K0: styles = w @ affine^T/sqrt(512) + bias ; sn = styles*rsqrt(mean(styles^2)) ---- */
__global__ void k0_styles(const float* __restrict__ w, const float* __restrict__ aw,
                          const float* __restrict__ ab, float* __restrict__ sn) {
    __shared__ float red[256];
    int c = threadIdx.x;
    const float inv = 0.044194173824159216f; /* 1/sqrt(512) */
    float st[BATCH];
    for (int b = 0; b < BATCH; ++b) {
        const float* wr = w + b*W_DIM;
        const float* ar = aw + (size_t)c*W_DIM;
        float s = 0.f;
        for (int d = 0; d < W_DIM; ++d) s += wr[d]*ar[d];
        st[b] = s*inv + ab[c];
    }
    float ss = 0.f;
    for (int b = 0; b < BATCH; ++b) ss += st[b]*st[b];
    red[c] = ss; __syncthreads();
    for (int o = 128; o > 0; o >>= 1) { if (c < o) red[c] += red[c+o]; __syncthreads(); }
    float rs = rsqrtf(red[0] / (float)(BATCH*IN_CH));
    for (int b = 0; b < BATCH; ++b) sn[b*IN_CH + c] = st[b]*rs;
}

/* ---- K1: per-o weight demod scale; write wsq[o][i] and bf16 weights pre-swizzled
        into exact B-fragment order [tap][kblk][nblk][lane][16] ---- */
__global__ void k1_wprep(const float* __restrict__ cw, float* __restrict__ wsq,
                         __bf16* __restrict__ wq) {
    __shared__ float red[256];
    int o = blockIdx.x, i = threadIdx.x;
    float w9[9]; float ss = 0.f;
    const float* p = cw + ((size_t)o*IN_CH + i)*9;
#pragma unroll
    for (int t = 0; t < 9; ++t) { w9[t] = p[t]; ss += w9[t]*w9[t]; }
    red[i] = ss; __syncthreads();
    for (int s = 128; s > 0; s >>= 1) { if (i < s) red[i] += red[i+s]; __syncthreads(); }
    float wscale = rsqrtf(red[0] / (float)(IN_CH*9));
    wsq[o*IN_CH + i] = ss * wscale * wscale;
    int kblk = i >> 5, krel = i & 31;
    int khalf = krel >> 4, kk = krel & 15;
    int nblk = o >> 4, nrel = o & 15;
    int lane = khalf*16 + nrel;
#pragma unroll
    for (int t = 0; t < 9; ++t) {
        size_t idx = ((((size_t)t*8 + kblk)*16 + nblk)*32 + lane)*16 + kk;
        wq[idx] = f2bf(w9[t]*wscale);
    }
}

/* ---- K2: dcoefs[b][o] = rsqrt(sum_i wsq[o][i]*sn[b][i]^2 + 1e-8) ---- */
__global__ void k2_dcoefs(const float* __restrict__ wsq, const float* __restrict__ sn,
                          float* __restrict__ dco) {
    int b = blockIdx.x, o = threadIdx.x;
    const float* snb = sn + b*IN_CH;
    const float* wr  = wsq + (size_t)o*IN_CH;
    float s = 0.f;
    for (int i = 0; i < IN_CH; ++i) { float v = snb[i]; s += wr[i]*v*v; }
    dco[b*OUT_CH + o] = rsqrtf(s + 1e-8f);
}

/* ---- K3: xt[b][pix][ci] = bf16(x[b][ci][pix] * sn[b][ci])  (LDS transpose, coalesced) ---- */
__global__ void k3_xt(const float* __restrict__ x, const float* __restrict__ sn,
                      __bf16* __restrict__ xt) {
    __shared__ float tile[64][65];
    int b = blockIdx.z, i0 = blockIdx.y*64, p0 = blockIdx.x*64;
    int tid = threadIdx.x;
#pragma unroll
    for (int rep = 0; rep < 16; ++rep) {
        int lin = rep*256 + tid;
        int ch = lin >> 6, px = lin & 63;
        tile[ch][px] = x[((size_t)b*IN_CH + i0 + ch)*NPIX + p0 + px];
    }
    __syncthreads();
#pragma unroll
    for (int rep = 0; rep < 16; ++rep) {
        int lin = rep*256 + tid;
        int px = lin >> 6, ch = lin & 63;
        float v = tile[ch][px] * sn[b*IN_CH + i0 + ch];
        xt[(((size_t)b*NPIX) + p0 + px)*IN_CH + i0 + ch] = f2bf(v);
    }
}

/* ---- K4: implicit-GEMM 3x3 conv via V_WMMA_F32_16X16X32_BF16.
   Weight panels (8 kblk x 4 nblk = 32KB/tap, pre-swizzled to fragment order)
   are DMA'd into LDS by the Tensor Data Mover, double-buffered across taps,
   tracked with TENSORcnt. A fragments stream from global (b128 pairs).
   Block: 8 waves as 4(M)x2(N); wave tile 32x32 (2x2 wmma, 32 acc VGPRs). ---- */
__global__ void k4_conv_wmma(const __bf16* __restrict__ xt, const __bf16* __restrict__ wq,
                             const float* __restrict__ dco, const float* __restrict__ cbias,
                             float* __restrict__ y) {
    __shared__ __bf16 Bs[2][8*4*32*16];   /* 2 x 32KB tap panels */

    const int tid  = threadIdx.x;
    const int lane = tid & 31, wid = tid >> 5;
    const int waveM = wid & 3, waveN = wid >> 2;
    const int b  = blockIdx.z;
    const int n0 = blockIdx.y*64 + waveN*32;
    const int m0 = blockIdx.x*128 + waveM*32;
    const int nb0 = blockIdx.y*4;         /* first global nblk of this block */

    v8f zero = {};
    v8f acc[2][2];
    acc[0][0] = zero; acc[0][1] = zero; acc[1][0] = zero; acc[1][1] = zero;

    int  pp[2], qq[2]; bool mok[2];
#pragma unroll
    for (int mi = 0; mi < 2; ++mi) {
        int m = m0 + mi*16 + (lane & 15);
        mok[mi] = (m < PCONV);
        int ms = mok[mi] ? m : 0;
        pp[mi] = ms / CONV_OUT; qq[mi] = ms - pp[mi]*CONV_OUT;
    }
    const int kb0 = (lane < 16) ? 0 : 16;

    union Frag { v16bf v; uint4 q[2]; };

    /* TDM descriptor group1 is tap-invariant: data_size=2B, 2D tile:
       tile_dim0 = tensor_dim0 = 2048 elems (4 nblk * 32 lanes * 16), tile_dim1 = 8 kblk rows,
       tensor_dim0_stride = 8192 elems (full 16-nblk kblk row). */
    const u32x8 g1 = { 0x00010000u,          /* data_size=1 (2B) */
                       2048u << 16,          /* tensor_dim0[15:0] */
                       8u << 16,             /* tensor_dim1[15:0] */
                       2048u << 16,          /* tile_dim0 */
                       8u,                   /* tile_dim1 */
                       8192u,                /* tensor_dim0_stride[31:0] */
                       0u, 0u };
    const u32x4 gz = { 0u, 0u, 0u, 0u };

    auto issue_tdm = [&](int tap, int buf) {
        uint64_t ga = (uint64_t)(uintptr_t)wq +
                      ((uint64_t)(tap*128 + nb0) * 512u) * 2u;   /* byte addr of panel */
        uint32_t ldsOff = (uint32_t)(uintptr_t)&Bs[buf][0];
        u32x4 g0;
        g0[0] = 1u;                                   /* count=1 */
        g0[1] = ldsOff;                               /* lds_addr */
        g0[2] = (uint32_t)ga;                         /* global_addr[31:0] */
        g0[3] = (uint32_t)((ga >> 32) & 0x01FFFFFFu) | 0x80000000u; /* addr[56:32] | type=2 */
        asm volatile("tensor_load_to_lds %0, %1, %2, %3"
                     :: "s"(g0), "s"(g1), "s"(gz), "s"(gz) : "memory");
    };

    if (wid == 0) issue_tdm(0, 0);

    int buf = 0;
    for (int tap = 0; tap < 9; ++tap) {
        const int dh = tap/3 - 2, dw = tap%3 - 2;
        size_t rowoff[2]; bool ok[2];
#pragma unroll
        for (int mi = 0; mi < 2; ++mi) {
            int ih = pp[mi] + dh, iw = qq[mi] + dw;
            ok[mi] = mok[mi] && (unsigned)ih < (unsigned)IN_SIZE && (unsigned)iw < (unsigned)IN_SIZE;
            rowoff[mi] = ((size_t)b*NPIX + (size_t)(ih*IN_SIZE + iw)) * IN_CH;
        }
        if (ok[0]) __builtin_prefetch(xt + rowoff[0], 0, 1); /* global_prefetch_b8 */

        if (wid == 0) __builtin_amdgcn_s_wait_tensorcnt(0);   /* panel[buf] landed */
        __syncthreads();
        if (wid == 0 && tap + 1 < 9) issue_tdm(tap + 1, buf ^ 1);

#pragma unroll
        for (int kblk = 0; kblk < 8; ++kblk) {
            const int kbase = kblk*32 + kb0;
            Frag af[2], bf[2];
#pragma unroll
            for (int mi = 0; mi < 2; ++mi) {
                if (ok[mi]) {
                    const uint4* pa = (const uint4*)(xt + rowoff[mi] + kbase);
                    af[mi].q[0] = pa[0];   /* K kb0..kb0+7   */
                    af[mi].q[1] = pa[1];   /* K kb0+8..+15   */
                } else {
                    af[mi].q[0] = make_uint4(0,0,0,0);
                    af[mi].q[1] = make_uint4(0,0,0,0);
                }
            }
#pragma unroll
            for (int ni = 0; ni < 2; ++ni) {   /* B fragments from LDS (ds_load_b128 x2) */
                const uint4* pb = (const uint4*)&Bs[buf][(((kblk*4 + waveN*2 + ni)*32 + lane) << 4)];
                bf[ni].q[0] = pb[0]; bf[ni].q[1] = pb[1];
            }
#pragma unroll
            for (int mi = 0; mi < 2; ++mi)
#pragma unroll
                for (int ni = 0; ni < 2; ++ni)
                    acc[mi][ni] = __builtin_amdgcn_wmma_f32_16x16x32_bf16(
                        false, af[mi].v, false, bf[ni].v,
                        (short)0, acc[mi][ni], false, false);
        }
        __syncthreads();   /* all reads of panel[buf] done before it is overwritten */
        buf ^= 1;
    }

    /* epilogue: demod scale + bias, scatter to y[b][o][m] */
#pragma unroll
    for (int ni = 0; ni < 2; ++ni) {
        int n = n0 + ni*16 + (lane & 15);
        float sc = dco[b*OUT_CH + n];
        float bs = cbias[n];
        float* yo = y + ((size_t)b*OUT_CH + n)*PCONV;
#pragma unroll
        for (int mi = 0; mi < 2; ++mi) {
            int mb = m0 + mi*16 + ((lane >> 4) << 3);
#pragma unroll
            for (int r = 0; r < 8; ++r) {
                int m = mb + r;
                if (m < PCONV) yo[m] = acc[mi][ni][r]*sc + bs;
            }
        }
    }
}

/* ---- K5: fused up2 -> 12tap sep -> leaky/clamp -> 12tap sep -> down2.
        One 32x32 out tile per block, whole chain in LDS (~52KB). ---- */
__global__ void k5_filter(const float* __restrict__ y, const float* __restrict__ fu,
                          const float* __restrict__ fd, float* __restrict__ out) {
    __shared__ float Ys[43*44];
    __shared__ float Vs[74*45];
    __shared__ float Zs[74*75];
    __shared__ float Ds[32*75];
    const int tid = threadIdx.x;
    const int c = blockIdx.y, b = blockIdx.z;
    const int u0 = (blockIdx.x & 3)*32, v0 = (blockIdx.x >> 2)*32;
    float fk2[12], fdr[12];
#pragma unroll
    for (int j = 0; j < 12; ++j) { fk2[j] = fu[11-j]*2.0f; fdr[j] = fd[11-j]; }
    const float* yp = y + ((size_t)b*OUT_CH + c)*PCONV;
    const int iyb = u0 - 5, ixb = v0 - 5;
    const int zr0 = 2*u0, zc0 = 2*v0;

    for (int idx = tid; idx < 43*43; idx += 256) {
        int yy = idx/43, xx = idx - yy*43;
        int iy = iyb + yy, ix = ixb + xx;
        float v = 0.f;
        if ((unsigned)iy < (unsigned)CONV_OUT && (unsigned)ix < (unsigned)CONV_OUT)
            v = yp[iy*CONV_OUT + ix];
        Ys[yy*44 + xx] = v;
    }
    __syncthreads();
    for (int idx = tid; idx < 74*43; idx += 256) {
        int r = idx/43, xx = idx - r*43;
        int zr = zr0 + r;
        float acc = 0.f;
#pragma unroll
        for (int a = 0; a < 12; ++a) {
            int t = zr + a - 9;
            if (t & 1) continue;
            int iy = (t >> 1) - iyb;
            if ((unsigned)iy < 43u) acc += fk2[a]*Ys[iy*44 + xx];
        }
        Vs[r*45 + xx] = acc;
    }
    __syncthreads();
    for (int idx = tid; idx < 74*74; idx += 256) {
        int r = idx/74, cc = idx - r*74;
        int zc = zc0 + cc;
        float acc = 0.f;
#pragma unroll
        for (int d = 0; d < 12; ++d) {
            int t = zc + d - 9;
            if (t & 1) continue;
            int jx = (t >> 1) - ixb;
            if ((unsigned)jx < 43u) acc += fk2[d]*Vs[r*45 + jx];
        }
        float z = (acc < 0.f) ? acc*0.2f : acc;
        z *= 1.4142135623730951f;
        z = fminf(fmaxf(z, -256.f), 256.f);
        Zs[r*75 + cc] = z;
    }
    __syncthreads();
    for (int idx = tid; idx < 32*74; idx += 256) {
        int u = idx/74, cc = idx - u*74;
        float acc = 0.f;
#pragma unroll
        for (int a = 0; a < 12; ++a) acc += fdr[a]*Zs[(2*u + a)*75 + cc];
        Ds[u*75 + cc] = acc;
    }
    __syncthreads();
    for (int idx = tid; idx < 32*32; idx += 256) {
        int u = idx >> 5, v = idx & 31;
        float acc = 0.f;
#pragma unroll
        for (int d = 0; d < 12; ++d) acc += fdr[d]*Ds[u*75 + 2*v + d];
        out[((size_t)b*OUT_CH + c)*NPIX + (u0 + u)*IN_SIZE + (v0 + v)] = acc;
    }
}

extern "C" void kernel_launch(void* const* d_in, const int* in_sizes, int n_in,
                              void* d_out, int out_size, void* d_ws, size_t ws_size,
                              hipStream_t stream) {
    const float* x  = (const float*)d_in[0];
    const float* w  = (const float*)d_in[1];
    const float* aw = (const float*)d_in[2];
    const float* ab = (const float*)d_in[3];
    const float* cw = (const float*)d_in[4];
    const float* cb = (const float*)d_in[5];
    const float* fu = (const float*)d_in[6];
    const float* fd = (const float*)d_in[7];
    float* out = (float*)d_out;
    char* ws = (char*)d_ws;
    float*  sn   = (float*)(ws + WS_SN);
    float*  dco  = (float*)(ws + WS_DCOEF);
    float*  wsq  = (float*)(ws + WS_WSQ);
    __bf16* wq   = (__bf16*)(ws + WS_WQ);
    __bf16* xt   = (__bf16*)(ws + WS_XT);
    float*  ybuf = (float*)(ws + WS_Y);

    k0_styles<<<1, 256, 0, stream>>>(w, aw, ab, sn);
    k1_wprep<<<OUT_CH, 256, 0, stream>>>(cw, wsq, wq);
    k2_dcoefs<<<BATCH, 256, 0, stream>>>(wsq, sn, dco);
    k3_xt<<<dim3(NPIX/64, IN_CH/64, BATCH), 256, 0, stream>>>(x, sn, xt);
    k4_conv_wmma<<<dim3((PCONV + 127)/128, OUT_CH/64, BATCH), 256, 0, stream>>>(xt, wq, dco, cb, ybuf);
    k5_filter<<<dim3(16, OUT_CH, BATCH), 256, 0, stream>>>(ybuf, fu, fd, out);
}
